// NMT_73641509257478
// MI455X (gfx1250) — matrix operand order repaired and checked by reading, not compiled, not run on hardware
//
#include <hip/hip_runtime.h>
#include <hip/hip_bf16.h>
#include <cstdint>

// ---------------------------------------------------------------------------
// MI455X (gfx1250) NMT forward pass.
//   - All GEMMs via V_WMMA_F32_16X16X32_BF16 (bf16 A/B, f32 accumulate).
//   - Input gates LN(x@Wih^T) hoisted out of the scans as large GEMMs.
//   - k_wmma_gemm stages the shared A tile into LDS with
//     GLOBAL_LOAD_ASYNC_TO_LDS_B128 (ASYNCcnt, double buffered) -> 8x dedup
//     of A traffic + copy/compute overlap; fragments come from ds_load_b128.
//     ASYNCcnt waits are executed by all waves (scalar, no-op for
//     non-issuing waves) so the K-loop has no EXEC-mask juggling.
//   - Recurrence: persistent workgroup per (layer, dir, batch-half); h/c and
//     the 16x2048 gate tile live in LDS (~180KB of the 320KB WGP LDS);
//     LayerNorms are wave-local shuffle reductions (wave w owns batch row w).
// ---------------------------------------------------------------------------

typedef __bf16 bf16;
typedef bf16  v16bf __attribute__((ext_vector_type(16)));
typedef bf16  v8bf  __attribute__((ext_vector_type(8)));
typedef float v8f   __attribute__((ext_vector_type(8)));

__device__ __forceinline__ float sigmoidf_(float x) { return 1.0f / (1.0f + __expf(-x)); }

// ---- CDNA5 async copy global->LDS (VGLOBAL encoding: vdst = LDS byte addr,
// ---- vaddr = 64-bit global addr). Tracked by ASYNCcnt.
__device__ __forceinline__ void async_ld_b128(unsigned lds_off, const void* gaddr) {
  asm volatile("global_load_async_to_lds_b128 %0, %1, off"
               :: "v"(lds_off), "v"((unsigned long long)(uintptr_t)gaddr)
               : "memory");
}
template <int N> __device__ __forceinline__ void wait_asynccnt() {
#if __has_builtin(__builtin_amdgcn_s_wait_asynccnt)
  __builtin_amdgcn_s_wait_asynccnt(N);
#else
  asm volatile("s_wait_asynccnt %0" :: "i"(N) : "memory");
#endif
}

// ------------------------------- small kernels ------------------------------

__global__ void k_cast_bf16(const float* __restrict__ in, bf16* __restrict__ out, int n) {
  int i = blockIdx.x * blockDim.x + threadIdx.x;
  if (i < n) out[i] = (bf16)in[i];
}

__global__ void k_gather(const int* __restrict__ ids, const float* __restrict__ emb,
                         bf16* __restrict__ out, int n_tok, int E) {
  int i = blockIdx.x * blockDim.x + threadIdx.x;
  if (i >= n_tok * E) return;
  int tok = i / E, e = i - tok * E;
  out[i] = (bf16)emb[(long)ids[tok] * E + e];
}

// Row-wise LayerNorm in place: X[row][0..width) -> (x-m)/sqrt(v+eps)*g + b
__global__ __launch_bounds__(256) void k_ln_rows(float* __restrict__ X, int width,
                                                 const float* __restrict__ g,
                                                 const float* __restrict__ b) {
  __shared__ float s_sum[8], s_sq[8];
  float* row = X + (long)blockIdx.x * width;
  float s = 0.f, q = 0.f;
  for (int i = threadIdx.x; i < width; i += 256) { float v = row[i]; s += v; q += v * v; }
  for (int m = 16; m; m >>= 1) { s += __shfl_xor(s, m, 32); q += __shfl_xor(q, m, 32); }
  if ((threadIdx.x & 31) == 0) { s_sum[threadIdx.x >> 5] = s; s_sq[threadIdx.x >> 5] = q; }
  __syncthreads();
  float ts = 0.f, tq = 0.f;
#pragma unroll
  for (int i = 0; i < 8; ++i) { ts += s_sum[i]; tq += s_sq[i]; }
  float mean = ts / (float)width;
  float rstd = rsqrtf(tq / (float)width - mean * mean + 1e-5f);
  for (int i = threadIdx.x; i < width; i += 256)
    row[i] = (row[i] - mean) * rstd * g[i] + b[i];
}

// ------------------------------- WMMA GEMM ----------------------------------
// C(M,N) f32 = A(M,K) bf16 row-major  x  W(N,K)^T  (+ bias[n])
// grid = (N/128, M/16), 256 threads = 8 waves, one 16x16 tile per wave.
// A tile (16x32 per K-step, shared by all 8 waves) is staged into LDS with
// async-to-LDS b128 (double buffered, 2 issuing waves); B streams per wave.
// Fragments use contiguous K, k = 16*(lane>>4) + s, same order in A and B (A 16x32, B 32x16).
__global__ __launch_bounds__(256)
void k_wmma_gemm(const bf16* __restrict__ A, int lda,
                 const bf16* __restrict__ W, int ldb,
                 float* __restrict__ C, long ldc,
                 const float* __restrict__ bias, int K) {
  __shared__ bf16 sA[2][16 * 32];                    // 2KB double buffer
  const int lane = threadIdx.x & 31;
  const int wv   = threadIdx.x >> 5;
  const int m0 = blockIdx.y * 16;
  const int n0 = (blockIdx.x * 8 + wv) * 16;
  const int am = lane & 15, ak = (lane >> 4) * 16;   // A: lane = M row, 16 contiguous K halves
  const int bn = lane & 15, bk = (lane >> 4) * 16;   // B: lane = N col, 16 contiguous K halves
  const bf16* bP = W + (long)(n0 + bn) * ldb + bk;

  // async staging: threads 0..63 (waves 0-1) each copy 16B of the 1KB A chunk
  const int sr = threadIdx.x >> 2;                   // row 0..15
  const int sc = (threadIdx.x & 3) * 8;              // 8-half column group
  const bf16* aStage = A + (long)(m0 + sr) * lda + sc;
  auto stage = [&](int buf, int k0) {
    if (threadIdx.x < 64)
      async_ld_b128((unsigned)(uintptr_t)&sA[buf][sr * 32 + sc], aStage + k0);
  };

  v8f acc = {0.f, 0.f, 0.f, 0.f, 0.f, 0.f, 0.f, 0.f};
  const int nk = K / 32;
  stage(0, 0);
#pragma unroll 1
  for (int i = 0; i < nk; ++i) {
    const int k0 = i * 32;
    // Uniform branch; waits are scalar + executed by every wave (no-op for
    // the 6 non-issuing waves whose ASYNCcnt is 0) -> no EXEC-mask dance.
    if (i + 1 < nk) {
      stage((i + 1) & 1, k0 + 32);
      wait_asynccnt<1>();                            // chunk i landed, i+1 in flight
    } else {
      wait_asynccnt<0>();                            // drain
    }
    __syncthreads();                                 // publish LDS chunk i
    const bf16* aP = &sA[i & 1][am * 32 + ak];       // ds_load_b128 fragments
    union { v16bf v; v8bf h[2]; } af;
    af.h[0] = *(const v8bf*)(aP);
    af.h[1] = *(const v8bf*)(aP + 8);
    v16bf bfr = *(const v16bf*)(bP + k0);
    if (k0 + 32 < K) __builtin_prefetch(bP + k0 + 32, 0, 1);   // global_prefetch_b8
    acc = __builtin_amdgcn_wmma_f32_16x16x32_bf16(false, af.v, false, bfr,
                                                  (short)0, acc, false, false);
    __syncthreads();                                 // reads done before buffer reuse
  }
  float bs = bias ? bias[n0 + (lane & 15)] : 0.0f;
  float* cP = C + (long)(m0 + 8 * (lane >> 4)) * ldc + n0 + (lane & 15);
#pragma unroll
  for (int r = 0; r < 8; ++r) cP[(long)r * ldc] = acc[r] + bs;  // row = m0+8*(lane>>4)+r
}

// ------------------------- persistent LSTM recurrence -----------------------
// Block = 16 waves; block x = batch half (rows b0..b0+15). Wave w owns row w.
// Per step: hgates = h @ Whh^T (WMMA with A from LDS), wave-local LN(2048),
// add precomputed LN'ed input gates, pointwise, wave-local LN(512) on c.
__global__ __launch_bounds__(512)
void k_lstm_scan(const float* __restrict__ ig,   // (T*32, 2048) = LN(x@Wih^T)
                 const bf16*  __restrict__ Whh,  // (2048, 512) bf16
                 const float* __restrict__ gh, const float* __restrict__ bh,
                 const float* __restrict__ gc, const float* __restrict__ bc,
                 const bf16*  __restrict__ h0, const float* __restrict__ c0, // nullable (32,512)
                 bf16* __restrict__ y, int ldy, int col_off,                 // y row = t*32+b
                 bf16* __restrict__ hT, bf16* __restrict__ cT, int stT,      // nullable finals
                 int T, int reverse) {
  extern __shared__ char smem[];
  float* sh_gates = (float*)smem;                                  // 16*2048 f32 = 128KB
  bf16*  sh_h     = (bf16*)(smem + 16 * 2048 * 4);                 // 16*512 bf16 = 16KB
  float* sh_c     = (float*)(smem + 16 * 2048 * 4 + 16 * 512 * 2); // 16*512 f32  = 32KB
  const int lane  = threadIdx.x & 31;
  const int wv    = threadIdx.x >> 5;          // local batch row 0..15
  const int bglob = blockIdx.x * 16 + wv;

  // init state (wave-local rows)
  for (int j = 0; j < 16; ++j) {
    int u = lane + 32 * j;
    sh_h[wv * 512 + u] = h0 ? h0[(long)bglob * 512 + u] : (bf16)0.0f;
    sh_c[wv * 512 + u] = c0 ? c0[(long)bglob * 512 + u] : 0.0f;
  }

  const int am = lane & 15, ak = (lane >> 4) * 16;
  const int bn = lane & 15, bk = (lane >> 4) * 16;
  const bf16* aBase = sh_h + am * 512 + ak;     // A fragment from LDS (ds_load)
  const v8f vzero = {0.f, 0.f, 0.f, 0.f, 0.f, 0.f, 0.f, 0.f};

  for (int s = 0; s < T; ++s) {
    const int tt = reverse ? (T - 1 - s) : s;
    __syncthreads();   // h/c of prev step complete; gate buffer free

    // ---- hgates(16x2048) = h(16x512) @ Whh^T : wave wv covers cols [wv*128, wv*128+128)
    v8f acc[8];
#pragma unroll
    for (int i = 0; i < 8; ++i) acc[i] = vzero;
    const int n0w = wv * 128;
#pragma unroll 1
    for (int k0 = 0; k0 < 512; k0 += 32) {
      union { v16bf v; v8bf h[2]; } af;
      af.h[0] = *(const v8bf*)(aBase + k0);
      af.h[1] = *(const v8bf*)(aBase + k0 + 8);
#pragma unroll
      for (int i = 0; i < 8; ++i) {
        const bf16* bP = Whh + (long)(n0w + i * 16 + bn) * 512 + bk + k0;
        v16bf bfr = *(const v16bf*)bP;
        acc[i] = __builtin_amdgcn_wmma_f32_16x16x32_bf16(false, af.v, false, bfr,
                                                         (short)0, acc[i], false, false);
      }
    }
#pragma unroll
    for (int i = 0; i < 8; ++i) {
      float* gp = sh_gates + (8 * (lane >> 4)) * 2048 + n0w + i * 16 + (lane & 15);
#pragma unroll
      for (int r = 0; r < 8; ++r) gp[r * 2048] = acc[i][r];
    }
    __syncthreads();   // full gate rows visible

    // ---- wave-local LN stats over row wv (width 2048)
    const float* grow = sh_gates + wv * 2048;
    float s1 = 0.f, s2 = 0.f;
    for (int i = lane; i < 2048; i += 32) { float v = grow[i]; s1 += v; s2 += v * v; }
    for (int m = 16; m; m >>= 1) { s1 += __shfl_xor(s1, m, 32); s2 += __shfl_xor(s2, m, 32); }
    float gm = s1 * (1.0f / 2048.0f);
    float grs = rsqrtf(s2 * (1.0f / 2048.0f) - gm * gm + 1e-5f);

    // ---- pointwise: gates = LN(hgates) + ig ; c_pre
    const float* igrow = ig + (long)(tt * 32 + bglob) * 2048;
    float ovals[16];
#pragma unroll
    for (int j = 0; j < 16; ++j) {
      int u = lane + 32 * j;
      float gi_ = (grow[u       ] - gm) * grs * gh[u       ] + bh[u       ] + igrow[u       ];
      float gf_ = (grow[u +  512] - gm) * grs * gh[u +  512] + bh[u +  512] + igrow[u +  512];
      float gg_ = (grow[u + 1024] - gm) * grs * gh[u + 1024] + bh[u + 1024] + igrow[u + 1024];
      float go_ = (grow[u + 1536] - gm) * grs * gh[u + 1536] + bh[u + 1536] + igrow[u + 1536];
      ovals[j] = go_;
      float cp = sigmoidf_(gf_) * sh_c[wv * 512 + u] + sigmoidf_(gi_) * tanhf(gg_);
      sh_c[wv * 512 + u] = cp;   // wave-local row, in-order LDS
    }
    // ---- wave-local LN on c (width 512)
    float c1 = 0.f, c2 = 0.f;
#pragma unroll
    for (int j = 0; j < 16; ++j) { float v = sh_c[wv * 512 + lane + 32 * j]; c1 += v; c2 += v * v; }
    for (int m = 16; m; m >>= 1) { c1 += __shfl_xor(c1, m, 32); c2 += __shfl_xor(c2, m, 32); }
    float cm = c1 * (1.0f / 512.0f);
    float crs = rsqrtf(c2 * (1.0f / 512.0f) - cm * cm + 1e-5f);

    bf16* yrow = y + (long)(tt * 32 + bglob) * ldy + col_off;
#pragma unroll
    for (int j = 0; j < 16; ++j) {
      int u = lane + 32 * j;
      float cn = (sh_c[wv * 512 + u] - cm) * crs * gc[u] + bc[u];
      sh_c[wv * 512 + u] = cn;
      float hn = sigmoidf_(ovals[j]) * tanhf(cn);
      sh_h[wv * 512 + u] = (bf16)hn;
      yrow[u] = (bf16)hn;
    }
  }

  if (hT) {  // interleaved [hf,hb] finals: element = b*2H + u*stT (+dir via base ptr)
    for (int j = 0; j < 16; ++j) {
      int u = lane + 32 * j;
      hT[(long)bglob * 1024 + (long)u * stT] = sh_h[wv * 512 + u];
      cT[(long)bglob * 1024 + (long)u * stT] = (bf16)sh_c[wv * 512 + u];
    }
  }
}

// ------------------------------- host side ----------------------------------

struct Cell {
  const float *Whh, *Wih, *bc, *bh, *bi, *gc, *gh, *gi;
};
static Cell getCell(void* const* d_in, int base) {
  Cell c;
  c.Whh = (const float*)d_in[base + 0]; c.Wih = (const float*)d_in[base + 1];
  c.bc  = (const float*)d_in[base + 2]; c.bh  = (const float*)d_in[base + 3];
  c.bi  = (const float*)d_in[base + 4]; c.gc  = (const float*)d_in[base + 5];
  c.gh  = (const float*)d_in[base + 6]; c.gi  = (const float*)d_in[base + 7];
  return c;
}

extern "C" void kernel_launch(void* const* d_in, const int* in_sizes, int n_in,
                              void* d_out, int out_size, void* d_ws, size_t ws_size,
                              hipStream_t stream) {
  // ---- input map (pytree / sorted-dict-key flattening of setup_inputs) ----
  const int*   src     = (const int*)d_in[0];    // (40,32)
  const int*   tgt     = (const int*)d_in[1];    // (40,32)
  const float* aff_b   = (const float*)d_in[2];  // (512)
  const float* aff_w   = (const float*)d_in[3];  // (512,1024)
  Cell d0  = getCell(d_in, 4);    // dec[0]  (in=512)
  Cell d1  = getCell(d_in, 12);   // dec[1]  (in=512)
  Cell e0b = getCell(d_in, 20);   // enc[0].bwd (in=512)
  Cell e0f = getCell(d_in, 28);   // enc[0].fwd
  Cell e1b = getCell(d_in, 36);   // enc[1].bwd (in=1024)
  Cell e1f = getCell(d_in, 44);   // enc[1].fwd
  const float* out_b   = (const float*)d_in[52]; // (32000)
  const float* out_w   = (const float*)d_in[53]; // (32000,512)
  const float* src_emb = (const float*)d_in[54]; // (32000,512)
  const float* tgt_emb = (const float*)d_in[55]; // (32000,512)

  // ---- workspace carve-up ----
  char* ws = (char*)d_ws;
  size_t off = 0;
  auto alloc = [&](size_t bytes) -> char* {
    char* p = ws + off; off += (bytes + 255) & ~(size_t)255; return p;
  };
  bf16* w_e0f_ih = (bf16*)alloc(2048 * 512 * 2);
  bf16* w_e0f_hh = (bf16*)alloc(2048 * 512 * 2);
  bf16* w_e0b_ih = (bf16*)alloc(2048 * 512 * 2);
  bf16* w_e0b_hh = (bf16*)alloc(2048 * 512 * 2);
  bf16* w_e1f_ih = (bf16*)alloc((size_t)2048 * 1024 * 2);
  bf16* w_e1f_hh = (bf16*)alloc(2048 * 512 * 2);
  bf16* w_e1b_ih = (bf16*)alloc((size_t)2048 * 1024 * 2);
  bf16* w_e1b_hh = (bf16*)alloc(2048 * 512 * 2);
  bf16* w_d0_ih  = (bf16*)alloc(2048 * 512 * 2);
  bf16* w_d0_hh  = (bf16*)alloc(2048 * 512 * 2);
  bf16* w_d1_ih  = (bf16*)alloc(2048 * 512 * 2);
  bf16* w_d1_hh  = (bf16*)alloc(2048 * 512 * 2);
  bf16* w_aff    = (bf16*)alloc(512 * 1024 * 2);
  bf16* w_out    = (bf16*)alloc((size_t)32000 * 512 * 2);
  bf16* xa    = (bf16*)alloc((size_t)1280 * 512 * 2);   // src embeddings
  bf16* ta    = (bf16*)alloc((size_t)1280 * 512 * 2);   // tgt embeddings
  bf16* xcat  = (bf16*)alloc((size_t)1280 * 1024 * 2);  // enc L0 output [fwd|bwd]
  bf16* ydump = (bf16*)alloc((size_t)1280 * 1024 * 2);  // enc L1 output (unused)
  bf16* ya    = (bf16*)alloc((size_t)1280 * 512 * 2);   // dec L0 output
  bf16* yb    = (bf16*)alloc((size_t)1280 * 512 * 2);   // dec L1 output
  float* igF  = (float*)alloc((size_t)1280 * 2048 * 4); // LN'ed input gates (fwd/shared)
  float* igB  = (float*)alloc((size_t)1280 * 2048 * 4); // LN'ed input gates (bwd)
  bf16* hs_all = (bf16*)alloc(2 * 32 * 1024 * 2);       // (L,B,2H) interleaved finals
  bf16* cs_all = (bf16*)alloc(2 * 32 * 1024 * 2);
  float* haff  = (float*)alloc(64 * 512 * 4);           // (L*B,512) dec init h
  float* caff  = (float*)alloc(64 * 512 * 4);           // (L*B,512) dec init c
  bf16*  haff_bf = (bf16*)alloc(64 * 512 * 2);

  auto cast = [&](const float* s, bf16* d, size_t n) {
    k_cast_bf16<<<(int)((n + 255) / 256), 256, 0, stream>>>(s, d, (int)n);
  };
  auto gemm = [&](const bf16* A, int lda, const bf16* W, int ldb, float* C, long ldc,
                  const float* bias, int M, int N, int K) {
    dim3 g(N / 128, M / 16);
    k_wmma_gemm<<<g, 256, 0, stream>>>(A, lda, W, ldb, C, ldc, bias, K);
  };
  auto ln = [&](float* X, int rows, int width, const float* g, const float* b) {
    k_ln_rows<<<rows, 256, 0, stream>>>(X, width, g, b);
  };
  const size_t scan_smem = 16 * 2048 * 4 + 16 * 512 * 2 + 16 * 512 * 4; // 180224B of 320KB WGP LDS
  auto scan = [&](const float* ig, const bf16* Whh, const Cell& c,
                  const bf16* h0, const float* c0,
                  bf16* y, int ldy, int coff, bf16* hT, bf16* cT, int stT, int rev) {
    k_lstm_scan<<<2, 512, scan_smem, stream>>>(ig, Whh, c.gh, c.bh, c.gc, c.bc,
                                               h0, c0, y, ldy, coff, hT, cT, stT, 40, rev);
  };

  // ---- 0) weights -> bf16 ----
  cast(e0f.Wih, w_e0f_ih, 2048 * 512);  cast(e0f.Whh, w_e0f_hh, 2048 * 512);
  cast(e0b.Wih, w_e0b_ih, 2048 * 512);  cast(e0b.Whh, w_e0b_hh, 2048 * 512);
  cast(e1f.Wih, w_e1f_ih, (size_t)2048 * 1024); cast(e1f.Whh, w_e1f_hh, 2048 * 512);
  cast(e1b.Wih, w_e1b_ih, (size_t)2048 * 1024); cast(e1b.Whh, w_e1b_hh, 2048 * 512);
  cast(d0.Wih,  w_d0_ih,  2048 * 512);  cast(d0.Whh,  w_d0_hh,  2048 * 512);
  cast(d1.Wih,  w_d1_ih,  2048 * 512);  cast(d1.Whh,  w_d1_hh,  2048 * 512);
  cast(aff_w,   w_aff,    512 * 1024);
  cast(out_w,   w_out,    (size_t)32000 * 512);

  // ---- 1) embeddings ----
  k_gather<<<(1280 * 512 + 255) / 256, 256, 0, stream>>>(src, src_emb, xa, 1280, 512);
  k_gather<<<(1280 * 512 + 255) / 256, 256, 0, stream>>>(tgt, tgt_emb, ta, 1280, 512);

  // ---- 2) encoder layer 0 (in=512, bidirectional) ----
  gemm(xa, 512, w_e0f_ih, 512, igF, 2048, nullptr, 1280, 2048, 512);
  ln(igF, 1280, 2048, e0f.gi, e0f.bi);
  gemm(xa, 512, w_e0b_ih, 512, igB, 2048, nullptr, 1280, 2048, 512);
  ln(igB, 1280, 2048, e0b.gi, e0b.bi);
  scan(igF, w_e0f_hh, e0f, nullptr, nullptr, xcat, 1024, 0,   hs_all + 0,     cs_all + 0,     2, 0);
  scan(igB, w_e0b_hh, e0b, nullptr, nullptr, xcat, 1024, 512, hs_all + 1,     cs_all + 1,     2, 1);

  // ---- 3) encoder layer 1 (in=1024, bidirectional) ----
  gemm(xcat, 1024, w_e1f_ih, 1024, igF, 2048, nullptr, 1280, 2048, 1024);
  ln(igF, 1280, 2048, e1f.gi, e1f.bi);
  gemm(xcat, 1024, w_e1b_ih, 1024, igB, 2048, nullptr, 1280, 2048, 1024);
  ln(igB, 1280, 2048, e1b.gi, e1b.bi);
  scan(igF, w_e1f_hh, e1f, nullptr, nullptr, ydump, 1024, 0,   hs_all + 32768 + 0, cs_all + 32768 + 0, 2, 0);
  scan(igB, w_e1b_hh, e1b, nullptr, nullptr, ydump, 1024, 512, hs_all + 32768 + 1, cs_all + 32768 + 1, 2, 1);

  // ---- 4) affine bridge: (L*B,1024) @ aff_w^T + aff_b ----
  gemm(hs_all, 1024, w_aff, 1024, haff, 512, aff_b, 64, 512, 1024);
  gemm(cs_all, 1024, w_aff, 1024, caff, 512, aff_b, 64, 512, 1024);
  cast(haff, haff_bf, 64 * 512);

  // ---- 5) decoder layer 0 (in=512) ----
  gemm(ta, 512, w_d0_ih, 512, igF, 2048, nullptr, 1280, 2048, 512);
  ln(igF, 1280, 2048, d0.gi, d0.bi);
  scan(igF, w_d0_hh, d0, haff_bf, caff, ya, 512, 0, nullptr, nullptr, 1, 0);

  // ---- 6) decoder layer 1 (in=512) ----
  gemm(ya, 512, w_d1_ih, 512, igF, 2048, nullptr, 1280, 2048, 512);
  ln(igF, 1280, 2048, d1.gi, d1.bi);
  scan(igF, w_d1_hh, d1, haff_bf + 32 * 512, caff + 32 * 512, yb, 512, 0, nullptr, nullptr, 1, 0);

  // ---- 7) output projection: (1280,512) @ out_w^T + out_b -> d_out ----
  gemm(yb, 512, w_out, 512, (float*)d_out, 32000, out_b, 1280, 32000, 512);

  (void)in_sizes; (void)n_in; (void)out_size; (void)ws_size;
}
